// LocalizedFiltering_9328668967834
// MI455X (gfx1250) — matrix operand order repaired and checked by reading, not compiled, not run on hardware
//
#include <hip/hip_runtime.h>
#include <hip/hip_bf16.h>

// ---------------------------------------------------------------------------
// LocalizedFiltering on gfx1250: two causal size-2 convs == two bf16 WMMA
// GEMMs with per-row "previous token" pointers, fused residual + RMSNorm.
// M-tile = 32 tokens per 512-thread block (2 x 8-wave groups).
// sched_group_barrier pins each K-step to [loads][wmma] so B fragments get
// distinct registers -> partial s_wait_loadcnt + cross-iteration overlap.
// ---------------------------------------------------------------------------

typedef __attribute__((ext_vector_type(16))) __bf16 v16bf;
typedef __attribute__((ext_vector_type(8)))  float  v8f;

#define SGB __builtin_amdgcn_sched_group_barrier
#define SGB_VMEM_READ 0x0020
#define SGB_WMMA      0x0008

union BFrag { v16bf v; uint4 q[2]; };

// A fragment (16x32, MxK, bf16): lane L -> row (L&15), K = kb+{0..7, 16..23},
// kb = (L>=16)?8:0.  rowk points at row start + k0 (16B aligned).
__device__ __forceinline__ v16bf load_frag_a(const __hip_bfloat16* rowk, int lane) {
  const int kb = (lane >> 4) << 3;             // 0 or 8
  const uint4* p = reinterpret_cast<const uint4*>(rowk + kb);
  BFrag f; f.q[0] = p[0]; f.q[1] = p[2];       // +0..7  and +16..23
  return f.v;
}

// B fragment (32x16, KxN, bf16): lane L -> col (L&15), K = kb+{0..7, 16..23},
// kb = (L>=16)?8:0.  rowk points at Wt[n] + k0 (Wt stored N-major, [N][K]).
__device__ __forceinline__ v16bf load_frag_b(const __hip_bfloat16* rowk, int lane) {
  const int kb = (lane >> 4) << 3;             // 0 or 8
  const uint4* p = reinterpret_cast<const uint4*>(rowk + kb);
  BFrag f; f.q[0] = p[0]; f.q[1] = p[2];       // +0..7 and +16..23
  return f.v;
}

__device__ __forceinline__ int find_seq(const int* __restrict__ start, int B, int g) {
  int s = 0;
  while (s < B - 1 && start[s + 1] <= g) ++s;
  return s;
}

// --- repack weights: w (N, Kin, 2) f32  ->  Wt (N, 2*Kin) bf16, tap0 | tap1 -
__global__ void lf_prep_w(const float* __restrict__ w,
                          __hip_bfloat16* __restrict__ Wt,
                          int N, int Kin) {
  const long long K2 = 2LL * Kin;
  const long long tot = (long long)N * K2;
  for (long long idx = blockIdx.x * (long long)blockDim.x + threadIdx.x;
       idx < tot; idx += (long long)gridDim.x * blockDim.x) {
    const int n = (int)(idx / K2);
    const int k = (int)(idx - (long long)n * K2);
    const float v = (k < Kin) ? w[((size_t)n * Kin + k) * 2 + 0]
                              : w[((size_t)n * Kin + (k - Kin)) * 2 + 1];
    Wt[idx] = __float2bfloat16(v);
  }
}

// --- inputs f32 -> bf16; also emit lf1 (last token row per sequence) --------
__global__ void lf_convert_inputs(const float* __restrict__ inputs,
                                  __hip_bfloat16* __restrict__ xbf,
                                  float* __restrict__ lf1out,
                                  const int* __restrict__ start,
                                  const int* __restrict__ lens,
                                  int B, int E, int total) {
  const int g = blockIdx.x;
  if (g >= total) return;
  const int s = find_seq(start, B, g);
  const bool last = (g - start[s] == lens[s] - 1);
  const float* src = inputs + (size_t)g * E;
  __hip_bfloat16* dst = xbf + (size_t)g * E;
  for (int c = threadIdx.x; c < E; c += blockDim.x) {
    const float v = src[c];
    dst[c] = __float2bfloat16(v);
    if (last) lf1out[(size_t)s * E + c] = v;
  }
}

// --- per-sequence boundary vectors for the first token ----------------------
// p0 = max_len - len:
//   x_prev(first)  = (p0==0) ? lf1_cache : 0
//   o1_prev(first) = (p0==0) ? lf2_cache : (p0==1) ? W1_0 @ lf1 + b1 : b1
__global__ void lf_boundaries(const float* __restrict__ lf1c,
                              const float* __restrict__ lf2c,
                              const float* __restrict__ w1,
                              const float* __restrict__ b1,
                              __hip_bfloat16* __restrict__ xprevb,
                              __hip_bfloat16* __restrict__ o1b,
                              const int* __restrict__ lens,
                              const int* __restrict__ pmaxlen,
                              int E, int H) {
  const int s = blockIdx.x;
  const int p0 = pmaxlen[0] - lens[s];
  for (int c = threadIdx.x; c < E; c += blockDim.x)
    xprevb[(size_t)s * E + c] =
        __float2bfloat16(p0 == 0 ? lf1c[(size_t)s * E + c] : 0.f);
  for (int n = threadIdx.x; n < H; n += blockDim.x) {
    float v;
    if (p0 == 0) {
      v = lf2c[(size_t)s * H + n];
    } else {
      v = b1[n];
      if (p0 == 1) {
        const float* wr = w1 + (size_t)n * E * 2;  // w1[n][i][tap]
        const float* xr = lf1c + (size_t)s * E;
        for (int i = 0; i < E; ++i) v += wr[2 * i] * xr[i];
      }
    }
    o1b[(size_t)s * H + n] = __float2bfloat16(v);
  }
}

// --- GEMM1: O1[g] = [x_prev ; x_cur] @ W1t^T + b1  (K=2E=4096, N=H=1024) ----
// 512-thread block = 32 tokens; waves 0..7 -> tokens [0,16), waves 8..15 ->
// tokens [16,32); each wave owns 128 output cols (8 N-tiles).
__global__ __launch_bounds__(512) void lf_gemm1(
    const __hip_bfloat16* __restrict__ xbf,
    const __hip_bfloat16* __restrict__ xprevb,
    const __hip_bfloat16* __restrict__ W1t,   // [H][2E] bf16
    const float* __restrict__ b1,
    __hip_bfloat16* __restrict__ O1bf,        // [total][H] bf16
    float* __restrict__ lf2out,               // [B][H] f32
    const int* __restrict__ start, const int* __restrict__ lens,
    int B, int E, int H, int total) {
  const int lane = threadIdx.x & 31;
  const int wave = threadIdx.x >> 5;          // 0..15
  const int waveM = wave >> 3;                // 0/1: which 16-token half
  const int gblk = blockIdx.x * 32;
  const int g0 = gblk + waveM * 16;
  const int K1 = 2 * E;

  __shared__ int sh_s[32], sh_last[32];
  if (threadIdx.x < 32) {
    int gg = gblk + (int)threadIdx.x;
    if (gg > total - 1) gg = total - 1;
    const int s = find_seq(start, B, gg);
    sh_s[threadIdx.x] = s;
    sh_last[threadIdx.x] = (gg - start[s] == lens[s] - 1) ? 1 : 0;
  }
  __syncthreads();

  const int mrow = lane & 15;
  int g = g0 + mrow; if (g > total - 1) g = total - 1;
  const int s = find_seq(start, B, g);
  const int p = g - start[s];
  const __hip_bfloat16* aPrev = (p == 0) ? (xprevb + (size_t)s * E)
                                         : (xbf + (size_t)(g - 1) * E);
  const __hip_bfloat16* aCur = xbf + (size_t)g * E;
  const int n0w = (wave & 7) * 128;

  v8f acc[8] = {};
  // Two K-halves: [0,E) reads x_prev, [E,2E) reads x_cur (no per-iter select).
#pragma unroll 1
  for (int half = 0; half < 2; ++half) {
    const __hip_bfloat16* abase = half ? aCur : aPrev;
    const int kw0 = half ? E : 0;
#pragma unroll 1
    for (int k0 = 0; k0 < E; k0 += 32) {
      const v16bf a = load_frag_a(abase + k0, lane);
      v16bf bf[8];
#pragma unroll
      for (int nt = 0; nt < 8; ++nt)
        bf[nt] = load_frag_b(
            W1t + (size_t)(n0w + nt * 16 + mrow) * K1 + kw0 + k0, lane);
#pragma unroll
      for (int nt = 0; nt < 8; ++nt)
        acc[nt] = __builtin_amdgcn_wmma_f32_16x16x32_bf16(
            false, a, false, bf[nt], (short)0, acc[nt], false, false);
      // Pin shape: 18 VMEM reads (A:2 + B:16) then 8 WMMAs.
      SGB(SGB_VMEM_READ, 18, 0);
      SGB(SGB_WMMA, 8, 0);
    }
  }

  const int rbase = (lane >> 4) * 8;
#pragma unroll
  for (int nt = 0; nt < 8; ++nt) {
    const int n = n0w + nt * 16 + mrow;
    const float bias = b1[n];
#pragma unroll
    for (int j = 0; j < 8; ++j) {
      const int ml = waveM * 16 + rbase + j;   // block-local row
      const int gg = gblk + ml;
      if (gg >= total) continue;
      const float v = acc[nt][j] + bias;
      O1bf[(size_t)gg * H + n] = __float2bfloat16(v);
      if (sh_last[ml]) lf2out[(size_t)sh_s[ml] * H + n] = v;
    }
  }
}

// --- GEMM2 + bias + residual + RMSNorm, fully fused per 32-token block ------
// 512-thread block; each 8-wave group covers all E=2048 cols for its 16
// tokens (16 N-tiles per wave), sum(out^2) reduced in LDS per row.
__global__ __launch_bounds__(512) void lf_gemm2(
    const __hip_bfloat16* __restrict__ O1bf,
    const __hip_bfloat16* __restrict__ o1b,
    const __hip_bfloat16* __restrict__ W2t,   // [E][2H] bf16
    const float* __restrict__ b2, const float* __restrict__ gamma,
    const float* __restrict__ inputs,         // residual, f32
    float* __restrict__ out,                  // lf_output region of d_out
    const int* __restrict__ start, const int* __restrict__ lens,
    int B, int E, int H, int total) {
  extern __shared__ float smem[];
  float* sh_b2 = smem;            // E
  float* sh_gm = smem + E;        // E
  float* ssrow = smem + 2 * E;    // 32

  const int lane = threadIdx.x & 31;
  const int wave = threadIdx.x >> 5;          // 0..15
  const int waveM = wave >> 3;                // 0/1
  const int gblk = blockIdx.x * 32;
  const int g0 = gblk + waveM * 16;
  const int K2 = 2 * H;

  for (int c = threadIdx.x; c < E; c += blockDim.x) {
    sh_b2[c] = b2[c];
    sh_gm[c] = gamma[c];
  }
  if (threadIdx.x < 32) ssrow[threadIdx.x] = 0.f;
  __syncthreads();

  const int mrow = lane & 15;
  int g = g0 + mrow; if (g > total - 1) g = total - 1;
  const int s = find_seq(start, B, g);
  const int p = g - start[s];
  const __hip_bfloat16* aPrev = (p == 0) ? (o1b + (size_t)s * H)
                                         : (O1bf + (size_t)(g - 1) * H);
  const __hip_bfloat16* aCur = O1bf + (size_t)g * H;
  const int n0w = (wave & 7) * 256;

  v8f acc[16] = {};
#pragma unroll 1
  for (int half = 0; half < 2; ++half) {
    const __hip_bfloat16* abase = half ? aCur : aPrev;
    const int kw0 = half ? H : 0;
#pragma unroll 1
    for (int k0 = 0; k0 < H; k0 += 32) {
      const v16bf a = load_frag_a(abase + k0, lane);
#pragma unroll
      for (int nb = 0; nb < 4; ++nb) {         // batches of 4 B-frags
        v16bf bf[4];
#pragma unroll
        for (int t = 0; t < 4; ++t) {
          const int n = n0w + (nb * 4 + t) * 16 + mrow;
          bf[t] = load_frag_b(W2t + (size_t)n * K2 + kw0 + k0, lane);
        }
#pragma unroll
        for (int t = 0; t < 4; ++t)
          acc[nb * 4 + t] = __builtin_amdgcn_wmma_f32_16x16x32_bf16(
              false, a, false, bf[t], (short)0, acc[nb * 4 + t], false, false);
      }
      // Pin shape: A(2)+batch0(8) loads, 4 WMMA, then 8 loads + 4 WMMA x3.
      SGB(SGB_VMEM_READ, 10, 0);
      SGB(SGB_WMMA, 4, 0);
      SGB(SGB_VMEM_READ, 8, 0);
      SGB(SGB_WMMA, 4, 0);
      SGB(SGB_VMEM_READ, 8, 0);
      SGB(SGB_WMMA, 4, 0);
      SGB(SGB_VMEM_READ, 8, 0);
      SGB(SGB_WMMA, 4, 0);
    }
  }

  // bias + residual + sum of squares (group-level RMS over all E cols)
  const int rbase = (lane >> 4) * 8;
#pragma unroll
  for (int j = 0; j < 8; ++j) {
    const int ml = waveM * 16 + rbase + j;     // block-local row
    const int gg = gblk + ml;
    const bool ok = (gg < total);
    float pj = 0.f;
#pragma unroll
    for (int nt = 0; nt < 16; ++nt) {
      const int n = n0w + nt * 16 + mrow;
      const float res = ok ? inputs[(size_t)gg * E + n] : 0.f;
      const float v = acc[nt][j] + sh_b2[n] + res;
      acc[nt][j] = v;
      pj += v * v;
    }
    if (ok) atomicAdd(&ssrow[ml], pj);
  }
  __syncthreads();

#pragma unroll
  for (int j = 0; j < 8; ++j) {
    const int ml = waveM * 16 + rbase + j;
    const int gg = gblk + ml;
    if (gg >= total) continue;
    const float r = rsqrtf(ssrow[ml] / (float)E + 1e-6f);
#pragma unroll
    for (int nt = 0; nt < 16; ++nt) {
      const int n = n0w + nt * 16 + mrow;
      out[(size_t)gg * E + n] = sh_gm[n] * acc[nt][j] * r;
    }
  }
}

// ---------------------------------------------------------------------------
extern "C" void kernel_launch(void* const* d_in, const int* in_sizes, int n_in,
                              void* d_out, int out_size, void* d_ws, size_t ws_size,
                              hipStream_t stream) {
  const float* inputs = (const float*)d_in[0];
  const float* lf1c   = (const float*)d_in[1];
  const float* lf2c   = (const float*)d_in[2];
  const float* w1     = (const float*)d_in[3];
  const float* b1     = (const float*)d_in[4];
  const float* w2     = (const float*)d_in[5];
  const float* b2     = (const float*)d_in[6];
  const float* gamma  = (const float*)d_in[7];
  const int*   start  = (const int*)d_in[8];
  const int*   lens   = (const int*)d_in[9];
  const int*   pmax   = (const int*)d_in[10];
  (void)n_in; (void)out_size; (void)ws_size;

  const int B = in_sizes[9];
  const int E = in_sizes[1] / B;      // 2048
  const int H = in_sizes[2] / B;      // 1024
  const int total = in_sizes[0] / E;  // 24576

  // workspace partition (256B aligned)
  char* ws = (char*)d_ws;
  size_t off = 0;
  auto take = [&](size_t bytes) {
    char* p = ws + off;
    off = (off + bytes + 255) & ~(size_t)255;
    return p;
  };
  __hip_bfloat16* W1t    = (__hip_bfloat16*)take((size_t)H * 2 * E * sizeof(__hip_bfloat16));
  __hip_bfloat16* W2t    = (__hip_bfloat16*)take((size_t)E * 2 * H * sizeof(__hip_bfloat16));
  __hip_bfloat16* xbf    = (__hip_bfloat16*)take((size_t)total * E * sizeof(__hip_bfloat16));
  __hip_bfloat16* O1bf   = (__hip_bfloat16*)take((size_t)total * H * sizeof(__hip_bfloat16));
  __hip_bfloat16* xprevb = (__hip_bfloat16*)take((size_t)B * E * sizeof(__hip_bfloat16));
  __hip_bfloat16* o1b    = (__hip_bfloat16*)take((size_t)B * H * sizeof(__hip_bfloat16));

  float* out    = (float*)d_out;
  float* lf1out = out + (size_t)total * E;
  float* lf2out = lf1out + (size_t)B * E;

  lf_prep_w<<<2048, 256, 0, stream>>>(w1, W1t, H, E);
  lf_prep_w<<<2048, 256, 0, stream>>>(w2, W2t, E, H);
  lf_convert_inputs<<<total, 256, 0, stream>>>(inputs, xbf, lf1out, start, lens, B, E, total);
  lf_boundaries<<<B, 256, 0, stream>>>(lf1c, lf2c, w1, b1, xprevb, o1b, lens, pmax, E, H);

  const int mtiles = (total + 31) / 32;
  lf_gemm1<<<mtiles, 512, 0, stream>>>(xbf, xprevb, W1t, b1, O1bf, lf2out,
                                       start, lens, B, E, H, total);
  const size_t smem = (size_t)(2 * E + 32) * sizeof(float);
  lf_gemm2<<<mtiles, 512, smem, stream>>>(O1bf, o1b, W2t, b2, gamma, inputs, out,
                                          start, lens, B, E, H, total);
}
